// old_Block_3719441678665
// MI455X (gfx1250) — matrix operand order repaired and checked by reading, not compiled, not run on hardware
//
#include <hip/hip_runtime.h>
#include <hip/hip_bf16.h>

// ---------------------------------------------------------------- types
typedef __bf16 bf16;
typedef bf16  bf16x8  __attribute__((ext_vector_type(8)));
typedef bf16  bf16x16 __attribute__((ext_vector_type(16)));
typedef float v8f     __attribute__((ext_vector_type(8)));

#define M_ROWS 18464   // B*N = 32*577
#define CDIM   768
#define HID    3072
#define NQKV   2304
#define BH     384     // B*H
#define NP     608     // padded token count (19 * 32)
#define X_ELEMS 14180352ULL  // 32*577*768

__device__ __forceinline__ bf16x16 cat16(bf16x8 lo, bf16x8 hi) {
  return __builtin_shufflevector(lo, hi, 0,1,2,3,4,5,6,7,8,9,10,11,12,13,14,15);
}

// A-matrix 16x32 bf16 fragment (ISA: lanes0-15 K=0..7,16..23; lanes16-31 K=8..15,24..31)
__device__ __forceinline__ bf16x16 load_frag_a(const bf16* base, int pitch) {
  const int lane = threadIdx.x & 31;
  const int m = lane & 15, kh = lane >> 4;
  const bf16* p = base + m * pitch + kh * 8;
  return cat16(*(const bf16x8*)p, *(const bf16x8*)(p + 16));
}
// B-matrix 32x16 bf16 fragment: lane holds col n=lane&15, K=kh*8..+7 and 16+kh*8..+7.
// Source is row-major W[n][k] (computes A @ W^T).
__device__ __forceinline__ bf16x16 load_frag_b(const bf16* base, int pitch) {
  const int lane = threadIdx.x & 31;
  const int n = lane & 15, kh = lane >> 4;
  const bf16* p = base + n * pitch + kh * 8;
  return cat16(*(const bf16x8*)p, *(const bf16x8*)(p + 16));
}

__device__ __forceinline__ v8f wmma_bf16(bf16x16 a, bf16x16 b, v8f c) {
  return __builtin_amdgcn_wmma_f32_16x16x32_bf16(false, a, false, b, (short)0, c, false, false);
}

// CDNA5 async copy global->LDS (16B per lane), tracked by ASYNCcnt.
// LDS offset = low 32 bits of the generic pointer (LDS aperture truncation, ISA 10.2).
__device__ __forceinline__ void async_cp16(void* lds_p, const bf16* g) {
  unsigned lo = (unsigned)(uintptr_t)lds_p;
  asm volatile("global_load_async_to_lds_b128 %0, %1, off" :: "v"(lo), "v"(g) : "memory");
}
__device__ __forceinline__ void wait_async0() {
  asm volatile("s_wait_asynccnt 0" ::: "memory");
}

// ---------------------------------------------------------------- fp32 -> bf16 convert
__global__ __launch_bounds__(256) void cvt_kernel(const float* __restrict__ s,
                                                  bf16* __restrict__ d, int n) {
  int i = blockIdx.x * 256 + threadIdx.x;
  if (i < n) d[i] = (bf16)s[i];
}

// ---------------------------------------------------------------- LayerNorm (row = 768)
__global__ __launch_bounds__(256) void ln_kernel(const float* __restrict__ x,
                                                 const float* __restrict__ g,
                                                 const float* __restrict__ bta,
                                                 bf16* __restrict__ out) {
  const int row = blockIdx.x, tid = threadIdx.x;
  const float* xr = x + (size_t)row * CDIM;
  float v[3], s = 0.f, s2 = 0.f;
#pragma unroll
  for (int i = 0; i < 3; ++i) { v[i] = xr[tid + 256 * i]; s += v[i]; s2 += v[i] * v[i]; }
#pragma unroll
  for (int off = 16; off > 0; off >>= 1) { s += __shfl_xor(s, off); s2 += __shfl_xor(s2, off); }
  __shared__ float a1[8], a2[8];
  __shared__ float mean_s, rstd_s;
  if ((tid & 31) == 0) { a1[tid >> 5] = s; a2[tid >> 5] = s2; }
  __syncthreads();
  if (tid == 0) {
    float t1 = 0.f, t2 = 0.f;
    for (int i = 0; i < 8; ++i) { t1 += a1[i]; t2 += a2[i]; }
    float mean = t1 * (1.0f / CDIM);
    float var  = t2 * (1.0f / CDIM) - mean * mean;
    mean_s = mean; rstd_s = rsqrtf(var + 1e-6f);
  }
  __syncthreads();
  const float mean = mean_s, rstd = rstd_s;
#pragma unroll
  for (int i = 0; i < 3; ++i) {
    int c = tid + 256 * i;
    out[(size_t)row * CDIM + c] = (bf16)((v[i] - mean) * rstd * g[c] + bta[c]);
  }
}

// ---------------------------------------------------------------- tiled WMMA GEMM: C = A @ W^T + epi
// A: [M][K] bf16 row-major, W: [N][K] bf16 row-major.
// Block 128 thr = 4 waves; tile 32x128; each wave owns a 32x32 quadrant (4 accumulators).
// Double-buffered async global->LDS staging: one barrier + one asynccnt wait per K-step.
enum { EPI_QKV = 0, EPI_ADD = 1, EPI_GELU = 2 };

template <int EPI>
__global__ __launch_bounds__(128) void gemm_kernel(
    const bf16* __restrict__ A, const bf16* __restrict__ W,
    const float* __restrict__ bias, const int K, const int nstride,
    const float* __restrict__ resid, float* __restrict__ outf,
    bf16* __restrict__ outb,
    bf16* __restrict__ qb, bf16* __restrict__ kb, bf16* __restrict__ vtb) {
  __shared__ bf16 lA[2][32 * 40];
  __shared__ bf16 lW[2][128 * 40];
  const int tid = threadIdx.x;
  const int wave = tid >> 5, lane = tid & 31;
  const int m0 = blockIdx.x * 32;
  const int n0 = blockIdx.y * 128;
  const int r  = tid >> 2;          // 0..31
  const int kk = (tid & 3) * 8;     // 0,8,16,24

  auto issue = [&](int k0, int buf) {
    async_cp16(&lA[buf][r * 40 + kk], &A[(size_t)(m0 + r) * K + k0 + kk]);
#pragma unroll
    for (int c = 0; c < 4; ++c) {
      const int n = r + 32 * c;
      async_cp16(&lW[buf][n * 40 + kk], &W[(size_t)(n0 + n) * K + k0 + kk]);
    }
  };

  v8f acc[2][2] = {};
  const int nsteps = K >> 5;
  issue(0, 0);
  wait_async0();
  __syncthreads();
  for (int s = 0; s < nsteps; ++s) {
    const int cur = s & 1;
    if (s + 1 < nsteps) issue((s + 1) * 32, cur ^ 1);
    bf16x16 a0 = load_frag_a(&lA[cur][0], 40);
    bf16x16 a1 = load_frag_a(&lA[cur][16 * 40], 40);
    bf16x16 b0 = load_frag_b(&lW[cur][(wave * 2 + 0) * 16 * 40], 40);
    bf16x16 b1 = load_frag_b(&lW[cur][(wave * 2 + 1) * 16 * 40], 40);
    acc[0][0] = wmma_bf16(a0, b0, acc[0][0]);
    acc[0][1] = wmma_bf16(a0, b1, acc[0][1]);
    acc[1][0] = wmma_bf16(a1, b0, acc[1][0]);
    acc[1][1] = wmma_bf16(a1, b1, acc[1][1]);
    wait_async0();
    __syncthreads();
  }

  const int mh = (lane >> 4) * 8;
#pragma unroll
  for (int ctl = 0; ctl < 2; ++ctl) {
    const int n = n0 + (wave * 2 + ctl) * 16 + (lane & 15);
    const float bn = bias[n];
    int part = 0, hh = 0, dd = 0;
    if constexpr (EPI == EPI_QKV) { part = n / 768; hh = (n >> 6) % 12; dd = n & 63; }
#pragma unroll
    for (int e = 0; e < 8; ++e) {
#pragma unroll
      for (int rt = 0; rt < 2; ++rt) {
        const float v = acc[rt][ctl][e] + bn;
        const int m = m0 + rt * 16 + mh + e;
        if constexpr (EPI == EPI_QKV) {
          const int bb = m / 577, tok = m % 577;
          const int bh = bb * 12 + hh;
          const bf16 bv = (bf16)v;
          if (part == 0)      qb[((size_t)bh * NP + tok) * 64 + dd] = bv;
          else if (part == 1) kb[((size_t)bh * NP + tok) * 64 + dd] = bv;
          else                vtb[((size_t)bh * 64 + dd) * NP + tok] = bv;
        } else if constexpr (EPI == EPI_ADD) {
          outf[(size_t)m * nstride + n] = v + resid[(size_t)m * nstride + n];
        } else { // EPI_GELU (exact)
          const float gl = 0.5f * v * (1.0f + erff(v * 0.70710678118f));
          outb[(size_t)m * nstride + n] = (bf16)gl;
        }
      }
    }
  }
}

// ---------------------------------------------------------------- fused attention
// grid (37 row-tiles of 16, 384 bh), block 128 thr = 4 waves.
__global__ __launch_bounds__(128) void attn_kernel(
    const bf16* __restrict__ qbuf, const bf16* __restrict__ kbuf,
    const bf16* __restrict__ vtbuf, const float* __restrict__ aw,
    bf16* __restrict__ ao) {
  __shared__ bf16  lQ[16 * 72];
  __shared__ float lS[16 * 608];
  __shared__ bf16  lP[16 * 616];
  const int tid = threadIdx.x, wave = tid >> 5, lane = tid & 31;
  const int bh = blockIdx.y;
  const int b = bh / 12, h = bh % 12;
  const int tokbase = blockIdx.x * 16;
  const bf16* qrow  = qbuf  + (size_t)bh * NP * 64;
  const bf16* krow  = kbuf  + (size_t)bh * NP * 64;
  const bf16* vtrow = vtbuf + (size_t)bh * 64 * NP;

  { // load 16x64 Q tile
    const int r = tid >> 3, c = (tid & 7) * 8;
    *(bf16x8*)&lQ[r * 72 + c] = *(const bf16x8*)&qrow[(size_t)(tokbase + r) * 64 + c];
  }
  __syncthreads();

  // phase 1: S = 0.125 * Q K^T  (ctiles over padded 592 cols)
  for (int ct = wave; ct < 37; ct += 4) {
    v8f acc = {};
#pragma unroll
    for (int ks = 0; ks < 2; ++ks) {
      bf16x16 a  = load_frag_a(&lQ[ks * 32], 72);
      bf16x16 bf = load_frag_b(&krow[(size_t)(ct * 16) * 64 + ks * 32], 64);
      acc = wmma_bf16(a, bf, acc);
    }
    const int n  = ct * 16 + (lane & 15);
    const int mh = (lane >> 4) * 8;
#pragma unroll
    for (int e = 0; e < 8; ++e) {
      const int rr = mh + e;
      lS[rr * 608 + n] = (n < 577) ? acc[e] * 0.125f : -1e30f;
    }
  }
  __syncthreads();

  // CLS-row attention reweighting (pre-softmax multiply), tile holding row 0 only
  if (tokbase == 0) {
    for (int c = tid; c < 577; c += 128) {
      const float f = (c == 0) ? 1.0f : (aw[(size_t)b * 576 + (c - 1)] * 0.1f + 0.9f);
      lS[c] *= f;
    }
  }
  __syncthreads();

  // softmax: 8 lanes per row
  {
    const int rr = tid >> 3, j = tid & 7;
    float mx = -1e30f;
    for (int c = j; c < 592; c += 8) mx = fmaxf(mx, lS[rr * 608 + c]);
    mx = fmaxf(mx, __shfl_xor(mx, 1));
    mx = fmaxf(mx, __shfl_xor(mx, 2));
    mx = fmaxf(mx, __shfl_xor(mx, 4));
    float sum = 0.f;
    for (int c = j; c < 592; c += 8) {
      float p = __expf(lS[rr * 608 + c] - mx);
      lS[rr * 608 + c] = p;
      sum += p;
    }
    sum += __shfl_xor(sum, 1);
    sum += __shfl_xor(sum, 2);
    sum += __shfl_xor(sum, 4);
    const float inv = 1.0f / sum;
    for (int c = j; c < 592; c += 8)       lP[rr * 616 + c] = (bf16)(lS[rr * 608 + c] * inv);
    for (int c = 592 + j; c < 608; c += 8) lP[rr * 616 + c] = (bf16)0.0f;
  }
  __syncthreads();

  // phase 2: O = P @ V  (V stored transposed -> B-frag pattern), wave owns dtile = wave
  v8f o = {};
  for (int ks = 0; ks < 19; ++ks) {
    bf16x16 a  = load_frag_a(&lP[ks * 32], 616);
    bf16x16 bf = load_frag_b(&vtrow[(size_t)(wave * 16) * NP + ks * 32], NP);
    o = wmma_bf16(a, bf, o);
  }
  const int d  = wave * 16 + (lane & 15);
  const int mh = (lane >> 4) * 8;
#pragma unroll
  for (int e = 0; e < 8; ++e) {
    const int t = tokbase + mh + e;
    if (t < 577) ao[((size_t)(b * 577 + t)) * CDIM + h * 64 + d] = (bf16)o[e];
  }
}

// ---------------------------------------------------------------- patch_attn = pre-softmax CLS row
__global__ __launch_bounds__(128) void patch_kernel(const bf16* __restrict__ qb,
                                                    const bf16* __restrict__ kb,
                                                    float* __restrict__ pout) {
  const int bh = blockIdx.x;
  const int b = bh / 12, h = bh % 12;
  __shared__ float q0[64];
  if (threadIdx.x < 64) q0[threadIdx.x] = (float)qb[((size_t)bh * NP) * 64 + threadIdx.x];
  __syncthreads();
  for (int j = threadIdx.x; j < 576; j += 128) {
    const bf16x8* kr8 = (const bf16x8*)(kb + ((size_t)bh * NP + 1 + j) * 64);
    float acc = 0.f;
#pragma unroll
    for (int c = 0; c < 8; ++c) {
      bf16x8 kv = kr8[c];
#pragma unroll
      for (int d = 0; d < 8; ++d) acc += q0[c * 8 + d] * (float)kv[d];
    }
    pout[(size_t)b * 6912 + h * 576 + j] = 0.125f * acc;
  }
}

// ---------------------------------------------------------------- launch
extern "C" void kernel_launch(void* const* d_in, const int* in_sizes, int n_in,
                              void* d_out, int out_size, void* d_ws, size_t ws_size,
                              hipStream_t stream) {
  (void)in_sizes; (void)n_in; (void)out_size; (void)ws_size;
  const float* x      = (const float*)d_in[0];
  const float* aw     = (const float*)d_in[1];
  const float* ln1w   = (const float*)d_in[2];
  const float* ln1b   = (const float*)d_in[3];
  const float* qkv_w  = (const float*)d_in[4];
  const float* qkv_b  = (const float*)d_in[5];
  const float* proj_w = (const float*)d_in[6];
  const float* proj_b = (const float*)d_in[7];
  const float* ln2w   = (const float*)d_in[8];
  const float* ln2b   = (const float*)d_in[9];
  const float* fc1_w  = (const float*)d_in[10];
  const float* fc1_b  = (const float*)d_in[11];
  const float* fc2_w  = (const float*)d_in[12];
  const float* fc2_b  = (const float*)d_in[13];

  float* xres = (float*)d_out;             // [M][768] final x
  float* pout = (float*)d_out + X_ELEMS;   // [32][12][576] patch_attn

  char* ws = (char*)d_ws;
  size_t off = 0;
  auto alloc = [&](size_t bytes) -> void* {
    void* p = ws + off;
    off += (bytes + 255) & ~(size_t)255;
    return p;
  };
  bf16* Wq  = (bf16*)alloc((size_t)NQKV * CDIM * 2);
  bf16* Wp  = (bf16*)alloc((size_t)CDIM * CDIM * 2);
  bf16* W1  = (bf16*)alloc((size_t)HID * CDIM * 2);
  bf16* W2  = (bf16*)alloc((size_t)CDIM * HID * 2);
  bf16* hb  = (bf16*)alloc((size_t)M_ROWS * CDIM * 2);
  bf16* qb  = (bf16*)alloc((size_t)BH * NP * 64 * 2);
  bf16* kb  = (bf16*)alloc((size_t)BH * NP * 64 * 2);
  bf16* vtb = (bf16*)alloc((size_t)BH * 64 * NP * 2);
  bf16* ao  = (bf16*)alloc((size_t)M_ROWS * CDIM * 2);
  bf16* g1  = (bf16*)alloc((size_t)M_ROWS * HID * 2);

  auto cvt = [&](const float* s, bf16* d, int n) {
    cvt_kernel<<<(n + 255) / 256, 256, 0, stream>>>(s, d, n);
  };
  cvt(qkv_w,  Wq, NQKV * CDIM);
  cvt(proj_w, Wp, CDIM * CDIM);
  cvt(fc1_w,  W1, HID * CDIM);
  cvt(fc2_w,  W2, CDIM * HID);

  // LN1
  ln_kernel<<<M_ROWS, 256, 0, stream>>>(x, ln1w, ln1b, hb);
  // QKV projection, scatter to q / k / v^T
  gemm_kernel<EPI_QKV><<<dim3(M_ROWS / 32, NQKV / 128), 128, 0, stream>>>(
      hb, Wq, qkv_b, CDIM, 0, nullptr, nullptr, nullptr, qb, kb, vtb);
  // fused attention
  attn_kernel<<<dim3(37, BH), 128, 0, stream>>>(qb, kb, vtb, aw, ao);
  // patch_attn output
  patch_kernel<<<BH, 128, 0, stream>>>(qb, kb, pout);
  // proj + residual -> d_out
  gemm_kernel<EPI_ADD><<<dim3(M_ROWS / 32, CDIM / 128), 128, 0, stream>>>(
      ao, Wp, proj_b, CDIM, CDIM, x, xres, nullptr, nullptr, nullptr, nullptr);
  // LN2
  ln_kernel<<<M_ROWS, 256, 0, stream>>>(xres, ln2w, ln2b, hb);
  // FC1 + exact GELU
  gemm_kernel<EPI_GELU><<<dim3(M_ROWS / 32, HID / 128), 128, 0, stream>>>(
      hb, W1, fc1_b, CDIM, HID, nullptr, nullptr, g1, nullptr, nullptr, nullptr);
  // FC2 + residual (in-place on d_out)
  gemm_kernel<EPI_ADD><<<dim3(M_ROWS / 32, CDIM / 128), 128, 0, stream>>>(
      g1, W2, fc2_b, HID, CDIM, xres, xres, nullptr, nullptr, nullptr, nullptr);
}
